// GPTModel_9242769622344
// MI455X (gfx1250) — matrix operand order repaired and checked by reading, not compiled, not run on hardware
//
#include <hip/hip_runtime.h>
#include <hip/hip_bf16.h>
#include <math.h>

typedef __attribute__((ext_vector_type(16))) __bf16 v16bf;
typedef __attribute__((ext_vector_type(8)))  float  v8f;
typedef __attribute__((ext_vector_type(4)))  int    v4i;

#if defined(__has_builtin)
#if __has_builtin(__builtin_amdgcn_global_load_async_to_lds_b128)
#define HAVE_ASYNC_LDS 1
#endif
#endif
#ifndef HAVE_ASYNC_LDS
#define HAVE_ASYNC_LDS 0
#endif

namespace {
constexpr int LL   = 12;
constexpr int Dm   = 1024;
constexpr int Hh   = 16;
constexpr int DHd  = 64;
constexpr int Vv   = 50257;
constexpr int Ss   = 512;
constexpr int Bb   = 4;
constexpr int DFFd = 4096;
constexpr int Mrows = Bb * Ss;      // 2048

constexpr int TM = 128, TN = 128, TK = 32;
constexpr int LSTR = 40;            // LDS row stride in bf16 elems (80B, 16B aligned, conflict-spread)

constexpr int FLAG_GELU   = 1;
constexpr int FLAG_STORET = 2;      // store bf16 output transposed as vT[b, n(=h*64+dh), s]
}

#if HAVE_ASYNC_LDS
typedef __attribute__((address_space(1))) v4i* gv4p;   // global v4i*
typedef __attribute__((address_space(3))) v4i* lv4p;   // LDS v4i*
__device__ __forceinline__ void wait_async0() {
#if __has_builtin(__builtin_amdgcn_s_wait_asynccnt)
  __builtin_amdgcn_s_wait_asynccnt(0);
#else
  asm volatile("s_wait_asynccnt 0x0" ::: "memory");
#endif
}
#endif

// ---------------------------------------------------------------------------
// Stage one 128x32 A tile and one 128x32 (n-major) B tile into LDS.
// Async path: GLOBAL_LOAD_ASYNC_TO_LDS_B128 (no VGPR staging, ASYNCcnt).
// OOB B rows are pre-zeroed once before the K loop and simply skipped here.
// ---------------------------------------------------------------------------
__device__ __forceinline__ void stage_tile(
    const __bf16* __restrict__ A, const __bf16* __restrict__ Bm,
    int lda, int ldb, int Ndim, int m0, int n0, int kOff, int tid,
    __bf16 (*At)[LSTR], __bf16 (*Bt)[LSTR])
{
#pragma unroll
  for (int c = 0; c < 2; ++c) {
    const int chunk = tid + c * 256;            // 0..511
    const int row   = chunk >> 2;
    const int col8  = (chunk & 3) * 8;          // bf16 elems, 16B granules
    const __bf16* aSrc = A + (size_t)(m0 + row) * lda + kOff + col8;
    const int nrow = n0 + row;
#if HAVE_ASYNC_LDS
    __builtin_amdgcn_global_load_async_to_lds_b128(
        (gv4p)aSrc, (lv4p)&At[row][col8], 0, 0);
    if (nrow < Ndim)
      __builtin_amdgcn_global_load_async_to_lds_b128(
          (gv4p)(Bm + (size_t)nrow * ldb + kOff + col8),
          (lv4p)&Bt[row][col8], 0, 0);
#else
    *(uint4*)&At[row][col8] = *(const uint4*)aSrc;
    if (nrow < Ndim)
      *(uint4*)&Bt[row][col8] = *(const uint4*)(Bm + (size_t)nrow * ldb + kOff + col8);
    __builtin_prefetch(aSrc + TK, 0, 1);        // global_prefetch_b8
#endif
  }
}

// ---------------------------------------------------------------------------
// Generic bf16 WMMA GEMM:  C[m,n] = sum_k A[m,k] * B[n,k]   (B given N-major,
// i.e. "transposed" layout, so fragment K runs are always contiguous).
// Double-buffered LDS, async global->LDS staging, 8 waves, 64x32 per wave.
// Optional bias (per n), residual add (fp32 [m,n]), GELU, f32 and/or bf16
// outputs, z-batched with (b,h) offset decomposition.
// ---------------------------------------------------------------------------
__global__ __launch_bounds__(256) void gemm_bf16_wmma(
    const __bf16* __restrict__ A, const __bf16* __restrict__ Bm,
    const float* __restrict__ bias, const float* __restrict__ resid,
    float* __restrict__ outF, __bf16* __restrict__ outB,
    int Ndim, int Kdim, int lda, int ldb, int ldc,
    int sAb, int sAh, int sBb, int sBh, int sOb, int sOh,
    int hcount, int flags)
{
  __shared__ __bf16 At[2][TM][LSTR];
  __shared__ __bf16 Bt[2][TN][LSTR];

  const int z  = blockIdx.z;
  const int bb = z / hcount, hh = z % hcount;
  A  += (size_t)bb * sAb + (size_t)hh * sAh;
  Bm += (size_t)bb * sBb + (size_t)hh * sBh;
  const size_t obase = (size_t)bb * sOb + (size_t)hh * sOh;

  const int n0   = blockIdx.x * TN;
  const int m0   = blockIdx.y * TM;
  const int tid  = threadIdx.x;
  const int lane = tid & 31;
  const int wave = tid >> 5;
  const int wm   = (wave >> 2) * 64;   // 0 / 64
  const int wn   = (wave & 3) * 32;    // 0..96
  const int half = lane >> 4;
  const int l16  = lane & 15;

  // ---- pre-zero OOB B rows (both buffers) so staging can skip them ----
  if (tid < TM) {
    if (n0 + tid >= Ndim) {
      const uint4 zz = {0u, 0u, 0u, 0u};
#pragma unroll
      for (int c = 0; c < 4; ++c) {
        *(uint4*)&Bt[0][tid][c * 8] = zz;
        *(uint4*)&Bt[1][tid][c * 8] = zz;
      }
    }
  }
  __syncthreads();

  const v8f vzero = {0.f,0.f,0.f,0.f,0.f,0.f,0.f,0.f};
  v8f acc[4][2];
#pragma unroll
  for (int mi = 0; mi < 4; ++mi)
#pragma unroll
    for (int ni = 0; ni < 2; ++ni) acc[mi][ni] = vzero;

  union Frag { uint4 u[2]; v16bf v; };

  const int kTiles = Kdim / TK;
  // prologue: fill buffer 0
  stage_tile(A, Bm, lda, ldb, Ndim, m0, n0, 0, tid, At[0], Bt[0]);

  for (int kt = 0; kt < kTiles; ++kt) {
    const int cur = kt & 1;
#if HAVE_ASYNC_LDS
    wait_async0();               // our async loads into buf[cur] have landed
#endif
    __syncthreads();             // all waves' loads visible; prev-buffer reads done
    if (kt + 1 < kTiles)
      stage_tile(A, Bm, lda, ldb, Ndim, m0, n0, (kt + 1) * TK, tid,
                 At[cur ^ 1], Bt[cur ^ 1]);

    // ---- fragments: contiguous K order (K = half*16 + s), same for A and B ----
    Frag bfr[2];
#pragma unroll
    for (int ni = 0; ni < 2; ++ni) {
      const int n = wn + ni * 16 + l16;
      bfr[ni].u[0] = *(const uint4*)&Bt[cur][n][half * 16];      // K = half*16 .. +7
      bfr[ni].u[1] = *(const uint4*)&Bt[cur][n][half * 16 + 8];  // K = half*16+8 .. +15
    }
    Frag af[4];
#pragma unroll
    for (int mi = 0; mi < 4; ++mi) {
      const int m = wm + mi * 16 + l16;
      af[mi].u[0] = *(const uint4*)&At[cur][m][half * 16];       // K = half*16 .. +7
      af[mi].u[1] = *(const uint4*)&At[cur][m][half * 16 + 8];   // K = half*16+8 .. +15
    }
#pragma unroll
    for (int mi = 0; mi < 4; ++mi)
#pragma unroll
      for (int ni = 0; ni < 2; ++ni)
        acc[mi][ni] = __builtin_amdgcn_wmma_f32_16x16x32_bf16(
            false, af[mi].v, false, bfr[ni].v, (short)0, acc[mi][ni], false, false);
  }

  // ---- epilogue: C/D layout: VGPR r -> M = r + half*8, N = lane&15 ----
  const float GC = 0.7978845608f;   // sqrt(2/pi)
#pragma unroll
  for (int mi = 0; mi < 4; ++mi) {
#pragma unroll
    for (int ni = 0; ni < 2; ++ni) {
      const int tn = n0 + wn + ni * 16 + l16;
      if (tn >= Ndim) continue;
      const int tmb = m0 + wm + mi * 16 + half * 8;
#pragma unroll
      for (int r = 0; r < 8; ++r) {
        const int m = tmb + r;
        float v = acc[mi][ni][r];
        if (bias) v += bias[tn];
        if (flags & FLAG_GELU) {
          const float x = v;
          v = 0.5f * x * (1.0f + tanhf(GC * (x + 0.044715f * x * x * x)));
        }
        if (resid) v += resid[(size_t)m * ldc + tn];
        if (outF) outF[obase + (size_t)m * ldc + tn] = v;
        if (outB) {
          if (flags & FLAG_STORET) {
            const int bI = m >> 9, sI = m & (Ss - 1);       // m = b*512 + s
            outB[(size_t)bI * Dm * Ss + (size_t)tn * Ss + sI] = (__bf16)v;
          } else {
            outB[obase + (size_t)m * ldc + tn] = (__bf16)v;
          }
        }
      }
    }
  }
}

// ---------------------------------------------------------------------------
// LayerNorm over D=1024, one block per row, bf16 output (GEMM feed).
// ---------------------------------------------------------------------------
__global__ __launch_bounds__(256) void layernorm_bf16(
    const float* __restrict__ x, const float* __restrict__ sc,
    const float* __restrict__ sh, __bf16* __restrict__ out)
{
  const int row = blockIdx.x;
  const float* xr = x + (size_t)row * Dm;
  __shared__ float red[256];
  const int t = threadIdx.x;
  float vals[4];
  float s = 0.f;
#pragma unroll
  for (int i = 0; i < 4; ++i) { float v = xr[t + i * 256]; vals[i] = v; s += v; }
  red[t] = s; __syncthreads();
  for (int off = 128; off; off >>= 1) { if (t < off) red[t] += red[t + off]; __syncthreads(); }
  const float mean = red[0] * (1.0f / Dm);
  __syncthreads();
  float s2 = 0.f;
#pragma unroll
  for (int i = 0; i < 4; ++i) { float d = vals[i] - mean; s2 += d * d; }
  red[t] = s2; __syncthreads();
  for (int off = 128; off; off >>= 1) { if (t < off) red[t] += red[t + off]; __syncthreads(); }
  const float rstd = rsqrtf(red[0] * (1.0f / Dm) + 1e-6f);
#pragma unroll
  for (int i = 0; i < 4; ++i) {
    const int c = t + i * 256;
    out[(size_t)row * Dm + c] = (__bf16)((vals[i] - mean) * rstd * sc[c] + sh[c]);
  }
}

// ---------------------------------------------------------------------------
// Causal softmax: scores f32 [b,h,q,k] -> attn bf16; applies 1/sqrt(DH) scale
// and the j>q mask. grid = (S, B*H), 256 threads, 2 elems/thread.
// ---------------------------------------------------------------------------
__global__ __launch_bounds__(256) void softmax_causal(
    const float* __restrict__ scores, __bf16* __restrict__ attn)
{
  const int q  = blockIdx.x;
  const int bh = blockIdx.y;
  const float* row = scores + ((size_t)bh * Ss + q) * Ss;
  __bf16* orow = attn + ((size_t)bh * Ss + q) * Ss;
  __shared__ float red[256];
  const int t = threadIdx.x;
  const float a0 = (t       <= q) ? row[t]       * 0.125f : -INFINITY;
  const float a1 = (t + 256 <= q) ? row[t + 256] * 0.125f : -INFINITY;
  red[t] = fmaxf(a0, a1); __syncthreads();
  for (int off = 128; off; off >>= 1) { if (t < off) red[t] = fmaxf(red[t], red[t + off]); __syncthreads(); }
  const float mx = red[0];
  __syncthreads();
  const float e0 = (t       <= q) ? __expf(a0 - mx) : 0.f;
  const float e1 = (t + 256 <= q) ? __expf(a1 - mx) : 0.f;
  red[t] = e0 + e1; __syncthreads();
  for (int off = 128; off; off >>= 1) { if (t < off) red[t] += red[t + off]; __syncthreads(); }
  const float inv = 1.0f / red[0];
  orow[t]       = (__bf16)(e0 * inv);
  orow[t + 256] = (__bf16)(e1 * inv);
}

// ---------------------------------------------------------------------------
// Embedding gather + positional add (fp32).
// ---------------------------------------------------------------------------
__global__ __launch_bounds__(256) void embed_kernel(
    const int* __restrict__ idx, const float* __restrict__ tok,
    const float* __restrict__ pos, float* __restrict__ x)
{
  const long e = (long)blockIdx.x * 256 + threadIdx.x;
  if (e >= (long)Mrows * Dm) return;
  const int m = (int)(e >> 10), d = (int)(e & (Dm - 1)), s = m & (Ss - 1);
  x[e] = tok[(size_t)idx[m] * Dm + d] + pos[(size_t)s * Dm + d];
}

// fp32 [K,N] -> bf16 transposed [N,K]  (writes coalesced)
__global__ __launch_bounds__(256) void tcvt_kernel(
    const float* __restrict__ src, __bf16* __restrict__ dst, int K, int N)
{
  const long e = (long)blockIdx.x * 256 + threadIdx.x;
  if (e >= (long)K * N) return;
  const int n = (int)(e / K), k = (int)(e % K);
  dst[e] = (__bf16)src[(size_t)k * N + n];
}

// fp32 -> bf16 elementwise
__global__ __launch_bounds__(256) void cvt_kernel(
    const float* __restrict__ src, __bf16* __restrict__ dst, long n)
{
  const long e = (long)blockIdx.x * 256 + threadIdx.x;
  if (e < n) dst[e] = (__bf16)src[e];
}

// ---------------------------------------------------------------------------
extern "C" void kernel_launch(void* const* d_in, const int* in_sizes, int n_in,
                              void* d_out, int out_size, void* d_ws, size_t ws_size,
                              hipStream_t stream) {
  (void)in_sizes; (void)n_in; (void)out_size; (void)ws_size;
  const int*   in_idx  = (const int*)  d_in[0];
  const float* tok_emb = (const float*)d_in[1];
  const float* pos_emb = (const float*)d_in[2];
  const float* Wq = (const float*)d_in[3];
  const float* Wk = (const float*)d_in[4];
  const float* Wv = (const float*)d_in[5];
  const float* Wo = (const float*)d_in[6];
  const float* bo = (const float*)d_in[7];
  const float* W1 = (const float*)d_in[8];
  const float* b1 = (const float*)d_in[9];
  const float* W2 = (const float*)d_in[10];
  const float* b2 = (const float*)d_in[11];
  const float* ln1_s = (const float*)d_in[12];
  const float* ln1_b = (const float*)d_in[13];
  const float* ln2_s = (const float*)d_in[14];
  const float* ln2_b = (const float*)d_in[15];
  const float* lnf_s = (const float*)d_in[16];
  const float* lnf_b = (const float*)d_in[17];
  float* logits = (float*)d_out;

  // ---- workspace bump allocator ----
  char* p = (char*)d_ws;
  auto alloc = [&](size_t bytes) -> char* {
    char* r = p; p += (bytes + 255) & ~(size_t)255; return r;
  };
  const size_t DD = (size_t)Dm * Dm;
  __bf16* WqT = (__bf16*)alloc(LL * DD * 2);
  __bf16* WkT = (__bf16*)alloc(LL * DD * 2);
  __bf16* WvT = (__bf16*)alloc(LL * DD * 2);
  __bf16* WoT = (__bf16*)alloc(LL * DD * 2);
  __bf16* W1T = (__bf16*)alloc((size_t)LL * Dm * DFFd * 2);
  __bf16* W2T = (__bf16*)alloc((size_t)LL * Dm * DFFd * 2);
  __bf16* embB = (__bf16*)alloc((size_t)Vv * Dm * 2);
  float*  x0  = (float*)alloc((size_t)Mrows * Dm * 4);
  float*  x1  = (float*)alloc((size_t)Mrows * Dm * 4);
  __bf16* hB  = (__bf16*)alloc((size_t)Mrows * Dm * 2);
  __bf16* qB  = (__bf16*)alloc((size_t)Mrows * Dm * 2);
  __bf16* kB  = (__bf16*)alloc((size_t)Mrows * Dm * 2);
  __bf16* vtB = (__bf16*)alloc((size_t)Mrows * Dm * 2);
  __bf16* ctxB = (__bf16*)alloc((size_t)Mrows * Dm * 2);
  __bf16* a1B = (__bf16*)alloc((size_t)Mrows * DFFd * 2);
  float*  scoresF = (float*)alloc((size_t)Bb * Hh * Ss * Ss * 4);
  __bf16* attnB   = (__bf16*)alloc((size_t)Bb * Hh * Ss * Ss * 2);

  // ---- one-time (per call) weight convert + transpose to [N][K] bf16 ----
  const int gDD  = (int)((DD + 255) / 256);
  const int gDFF = (int)(((size_t)Dm * DFFd + 255) / 256);
  for (int l = 0; l < LL; ++l) {
    tcvt_kernel<<<gDD,  256, 0, stream>>>(Wq + l * DD, WqT + l * DD, Dm, Dm);
    tcvt_kernel<<<gDD,  256, 0, stream>>>(Wk + l * DD, WkT + l * DD, Dm, Dm);
    tcvt_kernel<<<gDD,  256, 0, stream>>>(Wv + l * DD, WvT + l * DD, Dm, Dm);
    tcvt_kernel<<<gDD,  256, 0, stream>>>(Wo + l * DD, WoT + l * DD, Dm, Dm);
    tcvt_kernel<<<gDFF, 256, 0, stream>>>(W1 + (size_t)l * Dm * DFFd, W1T + (size_t)l * Dm * DFFd, Dm, DFFd);
    tcvt_kernel<<<gDFF, 256, 0, stream>>>(W2 + (size_t)l * Dm * DFFd, W2T + (size_t)l * Dm * DFFd, DFFd, Dm);
  }
  cvt_kernel<<<(int)(((size_t)Vv * Dm + 255) / 256), 256, 0, stream>>>(tok_emb, embB, (long)Vv * Dm);

  // ---- embeddings ----
  embed_kernel<<<(Mrows * Dm) / 256, 256, 0, stream>>>(in_idx, tok_emb, pos_emb, x0);

  const dim3 blk(256);
  const dim3 gProj(Dm / TN, Mrows / TM, 1);          // 8 x 16
  const dim3 gFF1(DFFd / TN, Mrows / TM, 1);         // 32 x 16
  const dim3 gScore(Ss / TN, Ss / TM, Bb * Hh);      // 4 x 4 x 64
  const dim3 gCtx(1, Ss / TM, Bb * Hh);              // 1 x 4 x 64

  for (int l = 0; l < LL; ++l) {
    // --- attention sublayer ---
    layernorm_bf16<<<Mrows, blk, 0, stream>>>(x0, ln1_s + l * Dm, ln1_b + l * Dm, hB);
    gemm_bf16_wmma<<<gProj, blk, 0, stream>>>(hB, WqT + l * DD, nullptr, nullptr, nullptr, qB,
        Dm, Dm, Dm, Dm, Dm, 0,0,0,0,0,0, 1, 0);
    gemm_bf16_wmma<<<gProj, blk, 0, stream>>>(hB, WkT + l * DD, nullptr, nullptr, nullptr, kB,
        Dm, Dm, Dm, Dm, Dm, 0,0,0,0,0,0, 1, 0);
    gemm_bf16_wmma<<<gProj, blk, 0, stream>>>(hB, WvT + l * DD, nullptr, nullptr, nullptr, vtB,
        Dm, Dm, Dm, Dm, Dm, 0,0,0,0,0,0, 1, FLAG_STORET);
    // scores[b,h,q,k] = q . k  (K = DH = 64)
    gemm_bf16_wmma<<<gScore, blk, 0, stream>>>(qB, kB, nullptr, nullptr, scoresF, nullptr,
        Ss, DHd, Dm, Dm, Ss,
        Ss * Dm, DHd, Ss * Dm, DHd, Hh * Ss * Ss, Ss * Ss, Hh, 0);
    softmax_causal<<<dim3(Ss, Bb * Hh), blk, 0, stream>>>(scoresF, attnB);
    // ctx[b,s,h,:] = attn . v   (K = S = 512, B-operand = vT contiguous-K)
    gemm_bf16_wmma<<<gCtx, blk, 0, stream>>>(attnB, vtB, nullptr, nullptr, nullptr, ctxB,
        DHd, Ss, Ss, Ss, Dm,
        Hh * Ss * Ss, Ss * Ss, Hh * DHd * Ss, DHd * Ss, Ss * Dm, DHd, Hh, 0);
    // x1 = x0 + ctx @ Wo + bo
    gemm_bf16_wmma<<<gProj, blk, 0, stream>>>(ctxB, WoT + l * DD, bo + l * Dm, x0, x1, nullptr,
        Dm, Dm, Dm, Dm, Dm, 0,0,0,0,0,0, 1, 0);
    // --- feed-forward sublayer ---
    layernorm_bf16<<<Mrows, blk, 0, stream>>>(x1, ln2_s + l * Dm, ln2_b + l * Dm, hB);
    gemm_bf16_wmma<<<gFF1, blk, 0, stream>>>(hB, W1T + (size_t)l * Dm * DFFd, b1 + (size_t)l * DFFd,
        nullptr, nullptr, a1B,
        DFFd, Dm, Dm, Dm, DFFd, 0,0,0,0,0,0, 1, FLAG_GELU);
    // x0 = x1 + a1 @ W2 + b2
    gemm_bf16_wmma<<<gProj, blk, 0, stream>>>(a1B, W2T + (size_t)l * Dm * DFFd, b2 + l * Dm,
        x1, x0, nullptr,
        Dm, DFFd, DFFd, DFFd, Dm, 0,0,0,0,0,0, 1, 0);
  }

  // ---- final LN + tied logits head ----
  layernorm_bf16<<<Mrows, blk, 0, stream>>>(x0, lnf_s, lnf_b, hB);
  gemm_bf16_wmma<<<dim3((Vv + TN - 1) / TN, Mrows / TM, 1), blk, 0, stream>>>(
      hB, embB, nullptr, nullptr, logits, nullptr,
      Vv, Dm, Dm, Dm, Vv, 0,0,0,0,0,0, 1, 0);
}
